// CustomGNN_5093831213713
// MI455X (gfx1250) — matrix operand order repaired and checked by reading, not compiled, not run on hardware
//
#include <hip/hip_runtime.h>

// ---------------------------------------------------------------------------
// Problem constants (match reference)
// ---------------------------------------------------------------------------
#define NN 32768      // nodes
#define DD 256        // feature dim
#define EE 524288     // edges
#define BG 128        // graphs
#define MX 384        // max nodes per graph
#define CS 32         // seeds
#define HH 4          // heads
#define DOUT 32

typedef __attribute__((ext_vector_type(16))) __bf16 v16bf;
typedef __attribute__((ext_vector_type(8)))  float  v8f;

union ABfrag { uint4 q[2]; v16bf v; };
union CFrag  { v8f v; float f[8]; };

__device__ __forceinline__ unsigned short f2bf(float x) {
  unsigned int u = __float_as_uint(x);
  u += 0x7FFFu + ((u >> 16) & 1u);           // round-to-nearest-even
  return (unsigned short)(u >> 16);
}

__device__ __forceinline__ float wave_sum(float v) {
#pragma unroll
  for (int o = 16; o > 0; o >>= 1) v += __shfl_xor(v, o, 32);
  return v;
}
__device__ __forceinline__ float wave_max(float v) {
#pragma unroll
  for (int o = 16; o > 0; o >>= 1) v = fmaxf(v, __shfl_xor(v, o, 32));
  return v;
}

// A fragment: rows m0..m0+15 (row-major, leading dim ld), K window [k0,k0+32)
// lane L elem e -> m = m0+(L&15), k = k0 + (e&8)*2 + ((L&16)>>1) + (e&7)
__device__ __forceinline__ v16bf frag_a(const unsigned short* __restrict__ base,
                                        int ld, int m0, int k0, int lane) {
  ABfrag f;
  const unsigned short* p = base + (m0 + (lane & 15)) * ld + k0 + ((lane & 16) >> 1);
  f.q[0] = *reinterpret_cast<const uint4*>(p);
  f.q[1] = *reinterpret_cast<const uint4*>(p + 16);
  return f.v;
}
// B fragment from B-transposed storage Bt[N][ld]: cols n0..n0+15, K window [k0,k0+32)
// lane L elem e -> n = n0+(L&15), k = k0 + (e&8)*2 + ((L&16)>>1) + (e&7)
__device__ __forceinline__ v16bf frag_b(const unsigned short* __restrict__ base,
                                        int ld, int n0, int k0, int lane) {
  ABfrag f;
  const unsigned short* p = base + (n0 + (lane & 15)) * ld + k0 + ((lane & 16) >> 1);
  f.q[0] = *reinterpret_cast<const uint4*>(p);
  f.q[1] = *reinterpret_cast<const uint4*>(p + 16);
  return f.v;
}

// ---------------------------------------------------------------------------
// Generic WMMA GEMM: C[M,N] = A[M,K] * Bt[N,K]^T (+bias), one 16x16 tile/wave.
// mode 0: outF = acc+bias           (f32)
// mode 1: outH = bf16(acc+bias)
// mode 2: outF = resid + relu(acc+bias)
// mode 3: outH[(g*256+col)*384+nn] = bf16(acc+bias), g=m/384 (per-graph V^T)
// ---------------------------------------------------------------------------
__global__ __launch_bounds__(256) void k_gemm(
    const unsigned short* __restrict__ A, const unsigned short* __restrict__ Bt,
    const float* __restrict__ bias, const float* __restrict__ resid,
    float* __restrict__ outF, unsigned short* __restrict__ outH,
    int N, int K, int mode) {
  int lane = threadIdx.x & 31;
  int tile = blockIdx.x * 8 + (threadIdx.x >> 5);
  int tn = N >> 4;
  int i = tile / tn, j = tile - i * tn;
  CFrag c;
#pragma unroll
  for (int r = 0; r < 8; ++r) c.f[r] = 0.0f;
  for (int k0 = 0; k0 < K; k0 += 32) {
    v16bf a = frag_a(A, K, i * 16, k0, lane);
    v16bf b = frag_b(Bt, K, j * 16, k0, lane);
    c.v = __builtin_amdgcn_wmma_f32_16x16x32_bf16(false, a, false, b,
                                                  (short)0, c.v, false, false);
  }
  int col = j * 16 + (lane & 15);
  int rb = i * 16 + ((lane >> 4) & 1) * 8;
  float bv = bias ? bias[col] : 0.0f;
#pragma unroll
  for (int r = 0; r < 8; ++r) {
    int m = rb + r;
    float v = c.f[r] + bv;
    if (mode == 0) {
      outF[m * N + col] = v;
    } else if (mode == 1) {
      outH[m * N + col] = f2bf(v);
    } else if (mode == 2) {
      outF[m * N + col] = resid[m * N + col] + fmaxf(v, 0.0f);
    } else {
      int g = m / 384, nn = m - g * 384;
      outH[(g * 256 + col) * 384 + nn] = f2bf(v);
    }
  }
}

// PMA scores: per (h,b): S[32,384] = q_h[32,64] * k_h[384,64]^T / 16
__global__ __launch_bounds__(256) void k_scores(
    const unsigned short* __restrict__ qb, const unsigned short* __restrict__ kb,
    float* __restrict__ sc) {
  int lane = threadIdx.x & 31;
  int hb = blockIdx.x / 6;
  int t = (blockIdx.x % 6) * 8 + (threadIdx.x >> 5);   // 0..47
  int i = t / 24, j = t - i * 24;
  int h = hb >> 7, b = hb & 127;
  const unsigned short* kB = kb + b * MX * DD;
  CFrag c;
#pragma unroll
  for (int r = 0; r < 8; ++r) c.f[r] = 0.0f;
#pragma unroll
  for (int k0 = 0; k0 < 64; k0 += 32) {
    v16bf a = frag_a(qb, DD, i * 16, h * 64 + k0, lane);
    v16bf bb = frag_b(kB, DD, j * 16, h * 64 + k0, lane);
    c.v = __builtin_amdgcn_wmma_f32_16x16x32_bf16(false, a, false, bb,
                                                  (short)0, c.v, false, false);
  }
  int col = j * 16 + (lane & 15);
  int rb = i * 16 + ((lane >> 4) & 1) * 8;
  float* o = sc + (size_t)hb * CS * MX;
#pragma unroll
  for (int r = 0; r < 8; ++r) o[(rb + r) * MX + col] = c.f[r] * 0.0625f;
}

// Softmax over 384 keys, one wave/row. Writes row-major probs (for A*V) and
// the permuted-transposed layout Abt[b2][n][h2*32+c] with h*128+b = b2*4+h2
// (replicates the reference's einops-on-concat quirk) for the comb GEMM.
__global__ __launch_bounds__(256) void k_softmax(
    const float* __restrict__ sc, const int* __restrict__ counts,
    unsigned short* __restrict__ Arow, unsigned short* __restrict__ Abt) {
  int lane = threadIdx.x & 31;
  int row = blockIdx.x * 8 + (threadIdx.x >> 5);   // ((h*128+b)*32 + c)
  int f = row >> 5;                                 // h*128+b
  int c = row & 31;
  int b = f & 127;
  int cnt = counts[b];
  float s[12];
  float mx = -3.4e38f;
#pragma unroll
  for (int p = 0; p < 12; ++p) {
    int n = lane + p * 32;
    float v = sc[(size_t)row * MX + n];
    if (n >= cnt) v -= 1e9f;
    s[p] = v;
    mx = fmaxf(mx, v);
  }
  mx = wave_max(mx);
  float sum = 0.0f;
#pragma unroll
  for (int p = 0; p < 12; ++p) { s[p] = __expf(s[p] - mx); sum += s[p]; }
  sum = wave_sum(sum);
  float inv = 1.0f / sum;
  int b2 = f >> 2, h2 = f & 3;
#pragma unroll
  for (int p = 0; p < 12; ++p) {
    int n = lane + p * 32;
    unsigned short pv = f2bf(s[p] * inv);
    Arow[(size_t)row * MX + n] = pv;
    Abt[((size_t)b2 * MX + n) * 128 + h2 * 32 + c] = pv;
  }
}

// Attention out: per (h,b): o[32,64] = q_h + A[32,384] * v_h[384,64]; concat heads.
__global__ __launch_bounds__(256) void k_attnout(
    const unsigned short* __restrict__ Arow, const unsigned short* __restrict__ vT,
    const float* __restrict__ qf, float* __restrict__ outp) {
  int lane = threadIdx.x & 31;
  int hb = blockIdx.x;
  int h = hb >> 7, b = hb & 127;
  int w = threadIdx.x >> 5;
  int i = w >> 2, j = w & 3;
  const unsigned short* Ar = Arow + (size_t)hb * CS * MX;
  const unsigned short* vt = vT + ((size_t)b * DD + h * 64) * MX;
  CFrag c;
#pragma unroll
  for (int r = 0; r < 8; ++r) c.f[r] = 0.0f;
  for (int k0 = 0; k0 < MX; k0 += 32) {
    v16bf a = frag_a(Ar, MX, i * 16, k0, lane);
    v16bf bb = frag_b(vt, MX, j * 16, k0, lane);
    c.v = __builtin_amdgcn_wmma_f32_16x16x32_bf16(false, a, false, bb,
                                                  (short)0, c.v, false, false);
  }
  int col = j * 16 + (lane & 15);
  int rb = i * 16 + ((lane >> 4) & 1) * 8;
#pragma unroll
  for (int r = 0; r < 8; ++r) {
    int cc = rb + r;
    int d = h * 64 + col;
    outp[((size_t)b * CS + cc) * DD + d] = qf[cc * DD + d] + c.f[r];
  }
}

// comb: per graph b: C[384,256] = Abt[b][384,128] * vns4t[b][256,128]^T; fused
// mean over 4 heads (x0.25), residual add with x_gcn, scatter to node rows.
__global__ __launch_bounds__(256) void k_comb(
    const unsigned short* __restrict__ Abt, const unsigned short* __restrict__ v4t,
    const int* __restrict__ node_of, const float* __restrict__ xg,
    float* __restrict__ xout) {
  int lane = threadIdx.x & 31;
  int b = blockIdx.x / 48;
  int t = (blockIdx.x % 48) * 8 + (threadIdx.x >> 5);  // 0..383
  int i = t / 16, j = t - i * 16;                      // i: n-tile(24), j: d-tile(16)
  const unsigned short* Ab = Abt + (size_t)b * MX * 128;
  const unsigned short* Vb = v4t + (size_t)b * DD * 128;
  CFrag c;
#pragma unroll
  for (int r = 0; r < 8; ++r) c.f[r] = 0.0f;
#pragma unroll
  for (int k0 = 0; k0 < 128; k0 += 32) {
    v16bf a = frag_a(Ab, 128, i * 16, k0, lane);
    v16bf bb = frag_b(Vb, 128, j * 16, k0, lane);
    c.v = __builtin_amdgcn_wmma_f32_16x16x32_bf16(false, a, false, bb,
                                                  (short)0, c.v, false, false);
  }
  int col = j * 16 + (lane & 15);
  int rb = i * 16 + ((lane >> 4) & 1) * 8;
#pragma unroll
  for (int r = 0; r < 8; ++r) {
    int n = rb + r;
    int node = node_of[b * MX + n];
    if (node >= 0)
      xout[(size_t)node * DD + col] = xg[(size_t)node * DD + col] + 0.25f * c.f[r];
  }
}

// Tiny exchanger attention (32 keys): one thread per (h,c), VALU.
__global__ __launch_bounds__(128) void k_exch_attn(
    const float* __restrict__ eq, const float* __restrict__ ek,
    const float* __restrict__ ev, float* __restrict__ outp) {
  int b = blockIdx.x;
  int h = threadIdx.x >> 5, c = threadIdx.x & 31;
  const float* qr = eq + ((size_t)b * CS + c) * DD + h * 64;
  float q[64];
#pragma unroll
  for (int k = 0; k < 64; ++k) q[k] = qr[k];
  float s[32];
  float mx = -3.4e38f;
  for (int n = 0; n < 32; ++n) {
    const float* kr = ek + ((size_t)b * CS + n) * DD + h * 64;
    float d = 0.0f;
#pragma unroll
    for (int k = 0; k < 64; ++k) d += q[k] * kr[k];
    d *= 0.0625f;
    s[n] = d;
    mx = fmaxf(mx, d);
  }
  float sum = 0.0f;
  for (int n = 0; n < 32; ++n) { s[n] = __expf(s[n] - mx); sum += s[n]; }
  float inv = 1.0f / sum;
  float o[64];
#pragma unroll
  for (int k = 0; k < 64; ++k) o[k] = q[k];
  for (int n = 0; n < 32; ++n) {
    float w = s[n] * inv;
    const float* vr = ev + ((size_t)b * CS + n) * DD + h * 64;
#pragma unroll
    for (int k = 0; k < 64; ++k) o[k] += w * vr[k];
  }
  float* orow = outp + ((size_t)b * CS + c) * DD + h * 64;
#pragma unroll
  for (int k = 0; k < 64; ++k) orow[k] = o[k];
}

// LayerNorm over 256 features, one wave per row.
__global__ __launch_bounds__(256) void k_layernorm(
    const float* __restrict__ in, float* __restrict__ outp,
    const float* __restrict__ g, const float* __restrict__ bta, int rows) {
  int row = blockIdx.x * 8 + (threadIdx.x >> 5);
  if (row >= rows) return;
  int lane = threadIdx.x & 31;
  const float* r = in + (size_t)row * DD;
  float v[8];
  float s = 0.0f;
#pragma unroll
  for (int p = 0; p < 8; ++p) { v[p] = r[lane + p * 32]; s += v[p]; }
  s = wave_sum(s) * (1.0f / 256.0f);
  float var = 0.0f;
#pragma unroll
  for (int p = 0; p < 8; ++p) { float d = v[p] - s; var += d * d; }
  var = wave_sum(var) * (1.0f / 256.0f);
  float inv = rsqrtf(var + 1e-5f);
  float* o = outp + (size_t)row * DD;
#pragma unroll
  for (int p = 0; p < 8; ++p) {
    int e = lane + p * 32;
    o[e] = (v[p] - s) * inv * g[e] + bta[e];
  }
}

// ------------------------- small utility kernels ---------------------------
__global__ void k_zero_u32(unsigned int* p, int n) {
  int g = blockIdx.x * blockDim.x + threadIdx.x;
  if (g < n) p[g] = 0u;
}
__global__ void k_fill_i32(int* p, int n, int v) {
  int g = blockIdx.x * blockDim.x + threadIdx.x;
  if (g < n) p[g] = v;
}
__global__ void k_copy_f4(const float4* __restrict__ a, float4* __restrict__ b, int n) {
  int g = blockIdx.x * blockDim.x + threadIdx.x;
  if (g < n) b[g] = a[g];
}
__global__ void k_cast_bf(const float* __restrict__ in, unsigned short* __restrict__ o, int n) {
  int g = blockIdx.x * blockDim.x + threadIdx.x;
  if (g < n) o[g] = f2bf(in[g]);
}
__global__ void k_transpose_w(const float* __restrict__ W, unsigned short* __restrict__ Wt,
                              int R, int C) {
  int g = blockIdx.x * blockDim.x + threadIdx.x;
  if (g < R * C) {
    int r = g / C, c = g - r * C;
    Wt[c * R + r] = f2bf(W[g]);
  }
}
__global__ void k_counts(const int* __restrict__ bid, int* __restrict__ counts, int n) {
  int g = blockIdx.x * blockDim.x + threadIdx.x;
  if (g < n) atomicAdd(&counts[bid[g]], 1);
}
__global__ void k_scan(const int* __restrict__ counts, int* __restrict__ starts) {
  if (threadIdx.x == 0 && blockIdx.x == 0) {
    int s = 0;
    for (int i = 0; i < BG; ++i) { starts[i] = s; s += counts[i]; }
  }
}
__global__ void k_pos(const int* __restrict__ bid, const int* __restrict__ starts,
                      int* __restrict__ pos, int* __restrict__ node_of, int n) {
  int g = blockIdx.x * blockDim.x + threadIdx.x;
  if (g < n) {
    int b = bid[g];
    int p = g - starts[b];
    pos[g] = p;
    node_of[b * MX + p] = g;
  }
}
__global__ void k_deg_edges(const int* __restrict__ dst, float* __restrict__ deg, int e) {
  int g = blockIdx.x * blockDim.x + threadIdx.x;
  if (g < e) atomicAdd(&deg[dst[g]], 1.0f);
}
__global__ void k_deg_fin(float* __restrict__ deg, float* __restrict__ dis, int n) {
  int g = blockIdx.x * blockDim.x + threadIdx.x;
  if (g < n) {
    float d = deg[g] + 1.0f;
    deg[g] = d;
    dis[g] = rsqrtf(d);
  }
}
__global__ void k_gcn_init(const float* __restrict__ xw, const float* __restrict__ deg,
                           const float* __restrict__ gb, float* __restrict__ xg, int n) {
  int g = blockIdx.x * blockDim.x + threadIdx.x;
  if (g < n) {
    int i = g >> 8, d = g & 255;
    xg[g] = xw[g] / deg[i] + gb[d];
  }
}
__global__ void k_gcn_edge(const int* __restrict__ src, const int* __restrict__ dst,
                           const float* __restrict__ dis, const float* __restrict__ xw,
                           float* __restrict__ xg) {
  int g = blockIdx.x * blockDim.x + threadIdx.x;   // E*64 threads, 4 dims each
  int e = g >> 6, q = (g & 63) << 2;
  int s = src[e], dd = dst[e];
  float c = dis[s] * dis[dd];
  const float* xs = xw + (size_t)s * DD + q;
  float* xd = xg + (size_t)dd * DD + q;
  atomicAdd(xd + 0, c * xs[0]);
  atomicAdd(xd + 1, c * xs[1]);
  atomicAdd(xd + 2, c * xs[2]);
  atomicAdd(xd + 3, c * xs[3]);
}
__global__ void k_dense_scatter(const float* __restrict__ xg, const int* __restrict__ bid,
                                const int* __restrict__ pos, unsigned short* __restrict__ dense,
                                int n) {
  int g = blockIdx.x * blockDim.x + threadIdx.x;
  if (g < n) {
    int i = g >> 8, d = g & 255;
    dense[((size_t)(bid[i] * MX + pos[i]) << 8) + d] = f2bf(xg[g]);
  }
}
// vns4t[b][d][h2*32+c] = bf16(vns[b][c][d])  (head-replicated transpose)
__global__ void k_vns4t(const float* __restrict__ vns, unsigned short* __restrict__ o, int n) {
  int g = blockIdx.x * blockDim.x + threadIdx.x;
  if (g < n) {
    int b = g >> 15;            // /(256*128)
    int rem = g & 32767;
    int d = rem >> 7;
    int c = rem & 31;
    o[g] = f2bf(vns[((size_t)b * CS + c) * DD + d]);
  }
}

// ---------------------------------------------------------------------------
extern "C" void kernel_launch(void* const* d_in, const int* in_sizes, int n_in,
                              void* d_out, int out_size, void* d_ws, size_t ws_size,
                              hipStream_t stream) {
  (void)in_sizes; (void)n_in; (void)out_size; (void)ws_size;
  const float* x_in    = (const float*)d_in[0];
  const float* gcn_W   = (const float*)d_in[1];
  const float* gcn_b   = (const float*)d_in[2];
  const float* seeds   = (const float*)d_in[3];
  const float* proj_W  = (const float*)d_in[4];
  const float* proj_b  = (const float*)d_in[5];
  const float* proj_lg = (const float*)d_in[6];
  const float* proj_lb = (const float*)d_in[7];
  const float* exch_W  = (const float*)d_in[8];
  const float* exch_b  = (const float*)d_in[9];
  const float* exch_lg = (const float*)d_in[10];
  const float* exch_lb = (const float*)d_in[11];
  const float* head_W  = (const float*)d_in[12];
  const float* head_b  = (const float*)d_in[13];
  const int*   eidx    = (const int*)d_in[14];
  const int*   bid     = (const int*)d_in[15];
  const int* esrc = eidx;
  const int* edst = eidx + EE;
  float* out = (float*)d_out;

  char* ws = (char*)d_ws;
  size_t off = 0;
  auto take = [&](size_t bytes) -> char* {
    char* p = ws + off;
    off += (bytes + 255) & ~(size_t)255;
    return p;
  };
  int*   counts  = (int*)take(BG * 4);
  int*   starts  = (int*)take(BG * 4);
  int*   pos     = (int*)take((size_t)NN * 4);
  int*   node_of = (int*)take((size_t)BG * MX * 4);
  float* deg     = (float*)take((size_t)NN * 4);
  float* dis     = (float*)take((size_t)NN * 4);
  float* x_cur   = (float*)take((size_t)NN * DD * 4);
  unsigned short* x_bf = (unsigned short*)take((size_t)NN * DD * 2);
  float* xw      = (float*)take((size_t)NN * DD * 4);
  float* x_gcn   = (float*)take((size_t)NN * DD * 4);
  unsigned short* dense_bf = (unsigned short*)take((size_t)BG * MX * DD * 2);
  unsigned short* wt       = (unsigned short*)take((size_t)DD * DD * 2);
  float* q_f     = (float*)take((size_t)CS * DD * 4);
  unsigned short* q_bf     = (unsigned short*)take((size_t)CS * DD * 2);
  unsigned short* seeds_bf = (unsigned short*)take((size_t)CS * DD * 2);
  unsigned short* k_bf     = (unsigned short*)take((size_t)BG * MX * DD * 2);
  unsigned short* vT_bf    = (unsigned short*)take((size_t)BG * DD * MX * 2);
  unsigned short* Arow     = (unsigned short*)take((size_t)HH * BG * CS * MX * 2);
  unsigned short* Abt      = (unsigned short*)take((size_t)BG * MX * 128 * 2);
  float* mab1    = (float*)take((size_t)BG * CS * DD * 4);
  float* t1      = (float*)take((size_t)BG * CS * DD * 4);
  float* t2      = (float*)take((size_t)BG * CS * DD * 4);
  unsigned short* tb       = (unsigned short*)take((size_t)BG * CS * DD * 2);
  float* vns     = (float*)take((size_t)BG * CS * DD * 4);
  unsigned short* vns_bf   = (unsigned short*)take((size_t)BG * CS * DD * 2);
  float* eq      = (float*)take((size_t)BG * CS * DD * 4);
  float* ek      = (float*)take((size_t)BG * CS * DD * 4);
  float* ev      = (float*)take((size_t)BG * CS * DD * 4);
  float* mab2    = (float*)take((size_t)BG * CS * DD * 4);
  unsigned short* v4t      = (unsigned short*)take((size_t)BG * DD * 128 * 2);
  float* scores = xw;   // alias: xw dead once scores are produced each layer

  auto gemm = [&](const unsigned short* A, const unsigned short* Bt, const float* bias,
                  const float* resid, float* oF, unsigned short* oH,
                  int M, int N, int K, int mode) {
    int blocks = (M >> 4) * (N >> 4) / 8;
    k_gemm<<<blocks, 256, 0, stream>>>(A, Bt, bias, resid, oF, oH, N, K, mode);
  };
  auto transW = [&](const float* W, int R, int C) {
    k_transpose_w<<<(R * C + 255) / 256, 256, 0, stream>>>(W, wt, R, C);
  };
  auto castbf = [&](const float* in, unsigned short* o, int n) {
    k_cast_bf<<<(n + 255) / 256, 256, 0, stream>>>(in, o, n);
  };
  auto lnorm = [&](const float* in, float* o, const float* g, const float* bb, int rows) {
    k_layernorm<<<rows / 8, 256, 0, stream>>>(in, o, g, bb, rows);
  };

  // -------- setup (per call; ws may be poisoned) --------
  k_zero_u32<<<1, 256, 0, stream>>>((unsigned*)counts, BG);
  k_counts<<<NN / 256, 256, 0, stream>>>(bid, counts, NN);
  k_scan<<<1, 32, 0, stream>>>(counts, starts);
  k_fill_i32<<<(BG * MX + 255) / 256, 256, 0, stream>>>(node_of, BG * MX, -1);
  k_pos<<<NN / 256, 256, 0, stream>>>(bid, starts, pos, node_of, NN);
  k_zero_u32<<<NN / 256, 256, 0, stream>>>((unsigned*)deg, NN);
  k_deg_edges<<<EE / 256, 256, 0, stream>>>(edst, deg, EE);
  k_deg_fin<<<NN / 256, 256, 0, stream>>>(deg, dis, NN);
  k_copy_f4<<<(NN * DD / 4) / 256, 256, 0, stream>>>((const float4*)x_in, (float4*)x_cur,
                                                     NN * DD / 4);
  k_zero_u32<<<(BG * MX * DD / 2) / 256, 256, 0, stream>>>((unsigned*)dense_bf,
                                                           BG * MX * DD / 2);

  for (int l = 0; l < 3; ++l) {
    // ---- GCN conv ----
    castbf(x_cur, x_bf, NN * DD);
    transW(gcn_W + (size_t)l * DD * DD, DD, DD);
    gemm(x_bf, wt, nullptr, nullptr, xw, nullptr, NN, DD, DD, 0);
    k_gcn_init<<<(NN * DD) / 256, 256, 0, stream>>>(xw, deg, gcn_b + l * DD, x_gcn, NN * DD);
    k_gcn_edge<<<(EE * 64) / 256, 256, 0, stream>>>(esrc, edst, dis, xw, x_gcn);

    // ---- to_dense_batch (bf16) ----
    k_dense_scatter<<<(NN * DD) / 256, 256, 0, stream>>>(x_gcn, bid, pos, dense_bf, NN * DD);

    // ---- PMA projections: q (shared across graphs), k, v^T ----
    castbf(seeds + (size_t)l * CS * DD, seeds_bf, CS * DD);
    transW(proj_W + (size_t)(l * 4 + 0) * DD * DD, DD, DD);
    gemm(seeds_bf, wt, proj_b + (l * 4 + 0) * DD, nullptr, q_f, nullptr, CS, DD, DD, 0);
    castbf(q_f, q_bf, CS * DD);
    transW(proj_W + (size_t)(l * 4 + 1) * DD * DD, DD, DD);
    gemm(dense_bf, wt, proj_b + (l * 4 + 1) * DD, nullptr, nullptr, k_bf, BG * MX, DD, DD, 1);
    transW(proj_W + (size_t)(l * 4 + 2) * DD * DD, DD, DD);
    gemm(dense_bf, wt, proj_b + (l * 4 + 2) * DD, nullptr, nullptr, vT_bf, BG * MX, DD, DD, 3);

    // ---- attention core (WMMA scores + softmax + WMMA A*V) ----
    k_scores<<<HH * BG * 6, 256, 0, stream>>>(q_bf, k_bf, scores);
    k_softmax<<<(HH * BG * CS) / 8, 256, 0, stream>>>(scores, counts, Arow, Abt);
    k_attnout<<<HH * BG, 256, 0, stream>>>(Arow, vT_bf, q_f, mab1);

    // ---- PMA tail: LN, FFN(+relu,+resid), LN ----
    lnorm(mab1, t1, proj_lg + (l * 2 + 0) * DD, proj_lb + (l * 2 + 0) * DD, BG * CS);
    castbf(t1, tb, BG * CS * DD);
    transW(proj_W + (size_t)(l * 4 + 3) * DD * DD, DD, DD);
    gemm(tb, wt, proj_b + (l * 4 + 3) * DD, t1, t2, nullptr, BG * CS, DD, DD, 2);
    lnorm(t2, vns, proj_lg + (l * 2 + 1) * DD, proj_lb + (l * 2 + 1) * DD, BG * CS);

    // ---- Exchanger SAB ----
    castbf(vns, vns_bf, BG * CS * DD);
    transW(exch_W + (size_t)(l * 4 + 0) * DD * DD, DD, DD);
    gemm(vns_bf, wt, exch_b + (l * 4 + 0) * DD, nullptr, eq, nullptr, BG * CS, DD, DD, 0);
    transW(exch_W + (size_t)(l * 4 + 1) * DD * DD, DD, DD);
    gemm(vns_bf, wt, exch_b + (l * 4 + 1) * DD, nullptr, ek, nullptr, BG * CS, DD, DD, 0);
    transW(exch_W + (size_t)(l * 4 + 2) * DD * DD, DD, DD);
    gemm(vns_bf, wt, exch_b + (l * 4 + 2) * DD, nullptr, ev, nullptr, BG * CS, DD, DD, 0);
    k_exch_attn<<<BG, 128, 0, stream>>>(eq, ek, ev, mab2);
    lnorm(mab2, t1, exch_lg + (l * 2 + 0) * DD, exch_lb + (l * 2 + 0) * DD, BG * CS);
    castbf(t1, tb, BG * CS * DD);
    transW(exch_W + (size_t)(l * 4 + 3) * DD * DD, DD, DD);
    gemm(tb, wt, exch_b + (l * 4 + 3) * DD, t1, t2, nullptr, BG * CS, DD, DD, 2);
    lnorm(t2, vns, exch_lg + (l * 2 + 1) * DD, exch_lb + (l * 2 + 1) * DD, BG * CS);

    // ---- comb einsum (WMMA) + residual scatter back to node features ----
    k_vns4t<<<(BG * DD * 128) / 256, 256, 0, stream>>>(vns, v4t, BG * DD * 128);
    k_comb<<<BG * 48, 256, 0, stream>>>(Abt, v4t, node_of, x_gcn, x_cur);
  }

  // ---- head: out = x @ head_W + head_b ----
  castbf(x_cur, x_bf, NN * DD);
  transW(head_W, DD, DOUT);
  gemm(x_bf, wt, head_b, nullptr, out, nullptr, NN, DOUT, DD, 0);
}
